// LSTMClass_58892591563431
// MI455X (gfx1250) — matrix operand order repaired and checked by reading, not compiled, not run on hardware
//
#include <hip/hip_runtime.h>
#include <stdint.h>

// ---- CDNA5 WMMA types ----
typedef __attribute__((ext_vector_type(16))) __bf16 v16bf;
typedef __attribute__((ext_vector_type(8)))  __bf16 v8bf;
typedef __attribute__((ext_vector_type(8)))  float  v8f;
typedef __attribute__((ext_vector_type(4)))  float  f32x4;

#define CAT16(lo, hi) __builtin_shufflevector(lo, hi, 0,1,2,3,4,5,6,7,8,9,10,11,12,13,14,15)

#define B_   32
#define T_   512
#define IN_  1024
#define H_   1024
#define G4_  4096
#define NWG_ 32          // persistent workgroups in phase 2

// ---- workspace layout (bytes) ----
static constexpr size_t WIH_OFF  = 0;                          //  8 MB bf16 W_ih
static constexpr size_t WHH_OFF  = (size_t)8  << 20;           //  8 MB bf16 W_hh
static constexpr size_t XB_OFF   = (size_t)16 << 20;           // 32 MB bf16 x
static constexpr size_t BIAS_OFF = (size_t)48 << 20;           // 16 KB f32 bias
static constexpr size_t HB0_OFF  = BIAS_OFF + (1u << 16);      // 64 KB bf16 h ping
static constexpr size_t HB1_OFF  = HB0_OFF  + (1u << 16);      // 64 KB bf16 h pong
static constexpr size_t C_OFF    = HB1_OFF  + (1u << 16);      // 128 KB f32 c
static constexpr size_t HF_OFF   = C_OFF    + (1u << 17);      // 128 KB f32 h shadow
static constexpr size_t BAR_OFF  = HF_OFF   + (1u << 17);      // grid barrier counter
static constexpr size_t XG_OFF   = (size_t)49 << 20;           // 128 MB bf16 xg [T][B][4H]

__device__ __forceinline__ float sigm(float x)  { return 1.f / (1.f + __expf(-x)); }
__device__ __forceinline__ float tanhf_(float x){ float e = __expf(-2.f * x); return (1.f - e) / (1.f + e); }

// monotonic grid-wide barrier (counter pre-zeroed by k_prep each call)
__device__ __forceinline__ void grid_sync(unsigned* bar, unsigned target)
{
    __syncthreads();                       // all block stores issued & drained
    if (threadIdx.x == 0) {
        __threadfence();                   // release: writes device-visible
        __hip_atomic_fetch_add(bar, 1u, __ATOMIC_RELAXED, __HIP_MEMORY_SCOPE_AGENT);
        while (__hip_atomic_load(bar, __ATOMIC_RELAXED, __HIP_MEMORY_SCOPE_AGENT) < target)
            __builtin_amdgcn_s_sleep(2);
        __threadfence();                   // acquire: drop stale WGP-local lines
    }
    __syncthreads();
}

// ------------------------------------------------------------------
// Prep: bf16 conversions, bias fold, state init, barrier reset
// ------------------------------------------------------------------
__global__ void k_prep(const float* __restrict__ x,  const float* __restrict__ h0,
                       const float* __restrict__ c0, const float* __restrict__ Wih,
                       const float* __restrict__ Whh, const float* __restrict__ bih,
                       const float* __restrict__ bhh,
                       __bf16* __restrict__ wih_b, __bf16* __restrict__ whh_b,
                       __bf16* __restrict__ x_b,   float* __restrict__ bias,
                       __bf16* __restrict__ h_b,   float* __restrict__ c,
                       float* __restrict__ hf,     unsigned* __restrict__ bar)
{
    long i = (long)blockIdx.x * blockDim.x + threadIdx.x;
    const long NW = (long)G4_ * IN_;          // 4M
    const long NX = (long)B_ * T_ * IN_;      // 16M
    if (i < NW) { wih_b[i] = (__bf16)Wih[i]; return; }
    i -= NW;
    if (i < NW) { whh_b[i] = (__bf16)Whh[i]; return; }
    i -= NW;
    if (i < NX) { x_b[i] = (__bf16)x[i]; return; }
    i -= NX;
    if (i < G4_) { bias[i] = bih[i] + bhh[i]; return; }
    i -= G4_;
    if (i < B_ * H_) { h_b[i] = (__bf16)h0[i]; hf[i] = h0[i]; return; }
    i -= B_ * H_;
    if (i < B_ * H_) { c[i] = c0[i]; return; }
    i -= B_ * H_;
    if (i == 0) { bar[0] = 0u; }
}

// ------------------------------------------------------------------
// Phase 1: xg[t*32+b][0..4095] = x[b][t][:] @ W_ih^T   (bf16 in, f32 acc, bf16 out)
// grid (128, 64), 256 thr: wave = one 16x64 tile, K=1024 loop
// ------------------------------------------------------------------
__global__ __launch_bounds__(256)
void k_xgemm(const __bf16* __restrict__ xb, const __bf16* __restrict__ wih,
             __bf16* __restrict__ xg)
{
    const int ln = threadIdx.x & 31;
    const int wv = threadIdx.x >> 5;            // 0..7
    const int rt = blockIdx.x * 8 + wv;         // row tile 0..1023
    const int row0 = rt * 16;
    const int nbase = blockIdx.y * 64;

    // A row: global row r = t*32+b  ->  x[b][t][:]
    const int r = row0 + (ln & 15);
    const int t = r >> 5, b = r & 31;
    const __bf16* arow = xb + ((long)(b * T_ + t)) * IN_;
    const int khalfA = (ln >> 4) * 8;           // A lane-half K offset
    const int khalfB = (ln >> 4) * 8;           // B lane-half K offset

    const __bf16* brow[4];
#pragma unroll
    for (int nt = 0; nt < 4; ++nt)
        brow[nt] = wih + (long)(nbase + nt * 16 + (ln & 15)) * IN_ + khalfB;

    v8f acc[4];
#pragma unroll
    for (int nt = 0; nt < 4; ++nt) acc[nt] = (v8f){0.f,0.f,0.f,0.f,0.f,0.f,0.f,0.f};

    for (int kb = 0; kb < IN_; kb += 32) {
        v8bf a0 = *(const v8bf*)(arow + kb + khalfA);
        v8bf a1 = *(const v8bf*)(arow + kb + khalfA + 16);
        v16bf A = CAT16(a0, a1);
#pragma unroll
        for (int nt = 0; nt < 4; ++nt) {
            v8bf b0 = *(const v8bf*)(brow[nt] + kb);
            v8bf b1 = *(const v8bf*)(brow[nt] + kb + 16);
            v16bf Bm = CAT16(b0, b1);
            acc[nt] = __builtin_amdgcn_wmma_f32_16x16x32_bf16(
                false, A, false, Bm, (short)0, acc[nt], false, false);
        }
    }

    const int hi = ln >> 4;
#pragma unroll
    for (int nt = 0; nt < 4; ++nt) {
        const int col = nbase + nt * 16 + (ln & 15);
#pragma unroll
        for (int vr = 0; vr < 8; ++vr) {
            const long rr = row0 + vr + 8 * hi;
            xg[rr * (long)G4_ + col] = (__bf16)acc[nt][vr];
        }
    }
}

// ------------------------------------------------------------------
// Phase 2: persistent LSTM scan. 32 WGs x 256 thr; WG owns 32 hidden units,
// loops t = 0..511 with a grid-wide barrier per step (no relaunch overhead).
// Wave q: gate = q>>1, sub = q&1 -> 16 gate columns, M=32 (2 tiles), K=1024.
// h_prev staged via GLOBAL_LOAD_ASYNC_TO_LDS_B128 (ASYNCcnt path).
// ------------------------------------------------------------------
__global__ __launch_bounds__(256)
void k_lstm(__bf16* __restrict__ hb0, __bf16* __restrict__ hb1,
            float* __restrict__ c, float* __restrict__ hf,
            const __bf16* __restrict__ whh, const __bf16* __restrict__ xg,
            const float* __restrict__ bias, unsigned* __restrict__ bar)
{
    __shared__ __align__(16) __bf16 hsh[32 * 512];   // 32 KB chunk of h_prev
    __shared__ float gbuf[4 * 32 * 32];              // 16 KB fused gate staging

    const int tid  = threadIdx.x;
    const int ln   = tid & 31;
    const int q    = tid >> 5;
    const int hb   = blockIdx.x * 32;                // hidden slice base
    const int gate = q >> 1, sub = q & 1;
    const int col0 = gate * H_ + hb + sub * 16;      // 16 gate columns of this wave

    const int khalfA = (ln >> 4) * 8;
    const int khalfB = (ln >> 4) * 8;
    const __bf16* brow = whh + (long)(col0 + (ln & 15)) * H_ + khalfB;
    const int hi = ln >> 4;
    const int n  = ln & 15;
    const float bsum = bias[col0 + n];

    for (int t = 0; t < T_; ++t) {
        const __bf16* hin  = (t & 1) ? hb1 : hb0;
        __bf16*       hout = (t & 1) ? hb0 : hb1;
        const __bf16* xg_t = xg + (long)t * B_ * G4_;

        v8f acc0 = (v8f){0.f,0.f,0.f,0.f,0.f,0.f,0.f,0.f};
        v8f acc1 = (v8f){0.f,0.f,0.f,0.f,0.f,0.f,0.f,0.f};

        for (int kc = 0; kc < H_; kc += 512) {
            // async stage of h_prev[:, kc:kc+512] directly into LDS (ASYNCcnt)
            for (int g = tid; g < 2048; g += 256) {
                const int e = g * 8;                  // 8 bf16 = 16 B per lane
                const int bb = e >> 9, kk = e & 511;
                const uint32_t ldsa = (uint32_t)(uintptr_t)(&hsh[e]);
                const uint64_t ga   = (uint64_t)(uintptr_t)(hin + bb * H_ + kc + kk);
                asm volatile("global_load_async_to_lds_b128 %0, %1, off"
                             :: "v"(ldsa), "v"(ga) : "memory");
            }
            asm volatile("s_wait_asynccnt 0x0" ::: "memory");
            __syncthreads();

            for (int kb = 0; kb < 512; kb += 32) {
                const int kg = kc + kb;
                v8bf b0 = *(const v8bf*)(brow + kg);
                v8bf b1 = *(const v8bf*)(brow + kg + 16);
                v16bf Bm = CAT16(b0, b1);

                const __bf16* ap0 = &hsh[(ln & 15) * 512 + kb + khalfA];
                v16bf A0 = CAT16(*(const v8bf*)ap0, *(const v8bf*)(ap0 + 16));
                acc0 = __builtin_amdgcn_wmma_f32_16x16x32_bf16(
                    false, A0, false, Bm, (short)0, acc0, false, false);

                const __bf16* ap1 = &hsh[((ln & 15) + 16) * 512 + kb + khalfA];
                v16bf A1 = CAT16(*(const v8bf*)ap1, *(const v8bf*)(ap1 + 16));
                acc1 = __builtin_amdgcn_wmma_f32_16x16x32_bf16(
                    false, A1, false, Bm, (short)0, acc1, false, false);
            }
            __syncthreads();
        }

        // add xg + bias, stage per-gate values in LDS
#pragma unroll
        for (int vr = 0; vr < 8; ++vr) {
            const int m0 = vr + 8 * hi;
            float v0 = acc0[vr] + bsum + (float)xg_t[(long)m0 * G4_ + col0 + n];
            gbuf[(gate * 32 + m0) * 32 + sub * 16 + n] = v0;
            const int m1 = m0 + 16;
            float v1 = acc1[vr] + bsum + (float)xg_t[(long)m1 * G4_ + col0 + n];
            gbuf[(gate * 32 + m1) * 32 + sub * 16 + n] = v1;
        }
        __syncthreads();

        // elementwise LSTM cell update for this WG's 32x32 (batch x hidden) patch
#pragma unroll
        for (int it = 0; it < 4; ++it) {
            const int idx = tid + it * 256;      // 0..1023
            const int bb = idx >> 5, j = idx & 31;
            const float ig = sigm (gbuf[(0 * 32 + bb) * 32 + j]);
            const float fg = sigm (gbuf[(1 * 32 + bb) * 32 + j]);
            const float gg = tanhf_(gbuf[(2 * 32 + bb) * 32 + j]);
            const float og = sigm (gbuf[(3 * 32 + bb) * 32 + j]);
            const int gj = hb + j;
            const float cp = c[bb * H_ + gj];
            const float cn = fg * cp + ig * gg;
            const float hn = og * tanhf_(cn);
            c[bb * H_ + gj]    = cn;
            hout[bb * H_ + gj] = (__bf16)hn;
            if (t == T_ - 1) hf[bb * H_ + gj] = hn;
        }

        grid_sync(bar, (unsigned)NWG_ * (unsigned)(t + 1));
    }
}

__global__ void k_out(const float* __restrict__ hf, float* __restrict__ out)
{
    const int i = blockIdx.x * 256 + threadIdx.x;
    if (i < B_ * H_) out[i] = hf[i];
}

// ------------------------------------------------------------------
extern "C" void kernel_launch(void* const* d_in, const int* in_sizes, int n_in,
                              void* d_out, int out_size, void* d_ws, size_t ws_size,
                              hipStream_t stream)
{
    const float* x   = (const float*)d_in[0];
    const float* h0  = (const float*)d_in[1];
    const float* c0  = (const float*)d_in[2];
    const float* Wih = (const float*)d_in[3];
    const float* Whh = (const float*)d_in[4];
    const float* bih = (const float*)d_in[5];
    const float* bhh = (const float*)d_in[6];

    char* ws = (char*)d_ws;
    __bf16*   wih_b = (__bf16*)  (ws + WIH_OFF);
    __bf16*   whh_b = (__bf16*)  (ws + WHH_OFF);
    __bf16*   xb    = (__bf16*)  (ws + XB_OFF);
    float*    bias  = (float*)   (ws + BIAS_OFF);
    __bf16*   hb0   = (__bf16*)  (ws + HB0_OFF);
    __bf16*   hb1   = (__bf16*)  (ws + HB1_OFF);
    float*    cbuf  = (float*)   (ws + C_OFF);
    float*    hfbuf = (float*)   (ws + HF_OFF);
    unsigned* bar   = (unsigned*)(ws + BAR_OFF);
    __bf16*   xg    = (__bf16*)  (ws + XG_OFF);

    // prep (also zeroes the grid barrier counter for this call)
    const long ntot = 2L * G4_ * IN_ + (long)B_ * T_ * IN_ + G4_ + 2L * B_ * H_ + 1;
    k_prep<<<(unsigned)((ntot + 255) / 256), 256, 0, stream>>>(
        x, h0, c0, Wih, Whh, bih, bhh, wih_b, whh_b, xb, bias, hb0, cbuf, hfbuf, bar);

    // phase 1: big input-projection GEMM
    dim3 ggrid(128, 64);
    k_xgemm<<<ggrid, 256, 0, stream>>>(xb, wih_b, xg);

    // phase 2: persistent recurrent scan (one launch, 512 grid barriers)
    k_lstm<<<NWG_, 256, 0, stream>>>(hb0, hb1, cbuf, hfbuf, whh_b, xg, bias, bar);

    // final h -> d_out (f32)
    k_out<<<(B_ * H_ + 255) / 256, 256, 0, stream>>>(hfbuf, (float*)d_out);
}
